// Block_69355131896526
// MI455X (gfx1250) — matrix operand order repaired and checked by reading, not compiled, not run on hardware
//
#include <hip/hip_runtime.h>
#include <hip/hip_bf16.h>

#define DEV __device__ __forceinline__

typedef __attribute__((ext_vector_type(16))) __bf16 bf16x16;
typedef __attribute__((ext_vector_type(8)))  float  f32x8;

union FragU { bf16x16 v; int4 q[2]; };

// ---------------- model constants ----------------
constexpr int BATCH   = 2;
constexpr int SEQ     = 2048;
constexpr int DMODEL  = 2048;
constexpr int DINNER  = 2048;
constexpr int NH      = 32;
constexpr int DST     = 64;   // d_state
constexpr int HD      = 64;   // head dim
constexpr int CHK     = 128;  // chunk length
constexpr int NCHK    = SEQ / CHK;          // 16
constexpr int CONVD   = DINNER + 2*NH*DST;  // 6144
constexpr int DINPROJ = 2*DINNER + 2*NH*DST + NH; // 8224
constexpr int DFF     = 8192;
constexpr int MTOT    = BATCH * SEQ;        // 4096

// ---------------- workspace layout ----------------
constexpr size_t A256(size_t x){ return (x + 255) & ~(size_t)255; }
constexpr size_t SZ_XN  = (size_t)MTOT * DMODEL * 2;          // bf16 normed act
constexpr size_t SZ_WBF = (size_t)DINPROJ * DMODEL * 2;       // largest bf16 weight
constexpr size_t SZ_ZX  = (size_t)MTOT * DINPROJ * 4;         // zxbcdt fp32
constexpr size_t SZ_XBC = (size_t)MTOT * CONVD * 4;           // conv output fp32
constexpr size_t SZ_CUM = (size_t)BATCH * NCHK * NH * CHK * 4;
constexpr size_t SZ_S   = (size_t)BATCH * NCHK * NH * DST * HD * 4;
constexpr size_t SZ_Y   = (size_t)MTOT * DINNER * 4;
constexpr size_t SZ_YBF = (size_t)MTOT * DINNER * 2;
constexpr size_t SZ_H1  = (size_t)MTOT * DMODEL * 4;
constexpr size_t SZ_G   = (size_t)MTOT * DFF * 2;

constexpr size_t OFF_XN  = 0;
constexpr size_t OFF_WBF = OFF_XN  + A256(SZ_XN);
constexpr size_t OFF_ZX  = OFF_WBF + A256(SZ_WBF);
constexpr size_t OFF_XBC = OFF_ZX  + A256(SZ_ZX);
constexpr size_t OFF_CUM = OFF_XBC + A256(SZ_XBC);
constexpr size_t OFF_S   = OFF_CUM + A256(SZ_CUM);
constexpr size_t OFF_HP  = OFF_S   + A256(SZ_S);
constexpr size_t OFF_Y   = OFF_HP  + A256(SZ_S);
constexpr size_t OFF_YBF = OFF_Y   + A256(SZ_Y);
constexpr size_t OFF_H1  = OFF_YBF + A256(SZ_YBF);
constexpr size_t OFF_H2N = OFF_H1  + A256(SZ_H1);
constexpr size_t OFF_G   = OFF_H2N + A256(SZ_XN);

// ---------------- helpers ----------------
DEV __bf16 f2bf(float f){
  union { float f; unsigned u; } c; c.f = f;
  unsigned r = (c.u + 0x7FFFu + ((c.u >> 16) & 1u)) >> 16;
  union { unsigned short s; __bf16 b; } o; o.s = (unsigned short)r;
  return o.b;
}
DEV float bf2f(__bf16 b){
  union { __bf16 b; unsigned short s; } c; c.b = b;
  union { unsigned u; float f; } o; o.u = (unsigned)c.s << 16;
  return o.f;
}
DEV float silu_f(float x){ return x / (1.f + __expf(-x)); }

DEV f32x8 wmma_bf16(bf16x16 a, bf16x16 b, f32x8 c){
  // D = A(16x32) * B(32x16) + C, fp32 accumulate
  return __builtin_amdgcn_wmma_f32_16x16x32_bf16(false, a, false, b, (short)0, c,
                                                 false, false);
}

// A-operand fragment (16x32 bf16): lane r holds row m0+r; group g selects K-halves
template<int S>
DEV bf16x16 frag_a(const __bf16 (*arr)[S], int m0, int kOff, int g, int r){
  const __bf16* row = arr[m0 + r];
  FragU u;
  u.q[0] = *reinterpret_cast<const int4*>(row + kOff + 8*g);
  u.q[1] = *reinterpret_cast<const int4*>(row + kOff + 16 + 8*g);
  return u.v;
}
// B-operand fragment (32x16 bf16) from [n][k]-major LDS: lane r = column n0+r
template<int S>
DEV bf16x16 frag_b(const __bf16 (*arr)[S], int n0, int kOff, int g, int r){
  const __bf16* row = arr[n0 + r];
  FragU u;
  u.q[0] = *reinterpret_cast<const int4*>(row + kOff + 8*g);
  u.q[1] = *reinterpret_cast<const int4*>(row + kOff + 16 + 8*g);
  return u.v;
}

// async global -> LDS 16B transfer (CDNA5 TDM-lite path, ASYNCcnt tracked)
DEV void async_b128(unsigned lds_addr, const void* gaddr){
  asm volatile("global_load_async_to_lds_b128 %0, %1, off"
               :: "v"(lds_addr), "v"((unsigned long long)(uintptr_t)gaddr)
               : "memory");
}
DEV void wait_async0(){
  asm volatile("s_wait_asynccnt 0" ::: "memory");
}

// ---------------- elementwise kernels ----------------
__global__ __launch_bounds__(256)
void f2b_kernel(const float* __restrict__ in, __bf16* __restrict__ out, int n){
  int i = blockIdx.x * 256 + threadIdx.x;
  if (i < n) out[i] = f2bf(in[i]);
}

__global__ __launch_bounds__(256)
void rmsnorm_kernel(const float* __restrict__ x, const float* __restrict__ w,
                    __bf16* __restrict__ out, int ncols){
  __shared__ float red[256];
  const int row = blockIdx.x;
  const float* xr = x + (size_t)row * ncols;
  float s = 0.f;
  for (int c = threadIdx.x; c < ncols; c += 256){ float v = xr[c]; s += v * v; }
  red[threadIdx.x] = s; __syncthreads();
  for (int off = 128; off > 0; off >>= 1){
    if (threadIdx.x < off) red[threadIdx.x] += red[threadIdx.x + off];
    __syncthreads();
  }
  const float inv = rsqrtf(red[0] / (float)ncols + 1e-5f);
  __bf16* orow = out + (size_t)row * ncols;
  for (int c = threadIdx.x; c < ncols; c += 256)
    orow[c] = f2bf(w[c] * xr[c] * inv);
}

// depthwise causal conv (K=4) over xBC slice of zxbcdt
__global__ __launch_bounds__(256)
void conv_kernel(const float* __restrict__ zx, const float* __restrict__ cw,
                 const float* __restrict__ cb, float* __restrict__ xbc){
  const int idx = blockIdx.x * 256 + threadIdx.x;   // b*SEQ*CONVD total, exact
  const int ch  = idx % CONVD;
  const int t   = (idx / CONVD) % SEQ;
  const int b   = idx / (CONVD * SEQ);
  const float* base = zx + (size_t)(b * SEQ) * DINPROJ + DINNER + ch;
  float acc = cb[ch];
#pragma unroll
  for (int k = 0; k < 4; k++){
    int tt = t - 3 + k;
    if (tt >= 0) acc += cw[ch * 4 + k] * base[(size_t)tt * DINPROJ];
  }
  xbc[idx] = acc;
}

// per (b,chunk,head): cum = cumsum(-softplus(dt)) over chunk
__global__ __launch_bounds__(128)
void cumsum_kernel(const float* __restrict__ zx, float* __restrict__ cum){
  __shared__ float arr[128];
  const int h = blockIdx.x & 31, cc = (blockIdx.x >> 5) & 15, b = blockIdx.x >> 9;
  const int i = threadIdx.x;
  const int pos = b * SEQ + cc * CHK + i;
  float dtv = zx[(size_t)pos * DINPROJ + DINNER + CONVD + h];
  float sp = (dtv > 20.f) ? dtv : log1pf(__expf(dtv));
  arr[i] = -sp;
  __syncthreads();
  for (int off = 1; off < 128; off <<= 1){
    float add = (i >= off) ? arr[i - off] : 0.f;
    __syncthreads();
    arr[i] += add;
    __syncthreads();
  }
  cum[(size_t)blockIdx.x * 128 + i] = arr[i];
}

// ---------------- WMMA GEMM: C[M,N] = A[M,K] * W[N,K]^T ----------------
// Async double-buffered LDS staging (global_load_async_to_lds_b128), K-slab = 64.
// EPI 0: out fp32      EPI 1: out bf16 = silu(c)
// EPI 2: out bf16 = c * bf2f(aux[idx])   EPI 3: out fp32 = c + res[idx]
template<int EPI>
__global__ __launch_bounds__(256)
void gemm_bf16_kernel(const __bf16* __restrict__ A, const __bf16* __restrict__ W,
                      void* __restrict__ outp, int M, int N, int K,
                      const float* __restrict__ res, const __bf16* __restrict__ aux){
  __shared__ alignas(16) __bf16 As[2][128][72];
  __shared__ alignas(16) __bf16 Ws[2][128][72];
  const int tid = threadIdx.x;
  const int w = tid >> 5, lane = tid & 31, g = lane >> 4, r = lane & 15;
  const int wrow = w >> 2, wcol = w & 3;          // 2 x 4 wave grid
  const int m0 = blockIdx.y * 128, n0 = blockIdx.x * 128;
  const int wm = wrow * 64, wn = wcol * 32;

  f32x8 acc[4][2];
#pragma unroll
  for (int i = 0; i < 4; i++) for (int j = 0; j < 2; j++) acc[i][j] = 0.f;

  // staging: thread -> (row, 32-half segment); 4x async b128 per operand per slab
  const int ldr = tid >> 1;            // tile row 0..127
  const int seg = (tid & 1) * 32;      // halves
  const __bf16* aRow = A + (size_t)(m0 + ldr) * K + seg;
  int wr = n0 + ldr; if (wr >= N) wr = N - 1;   // clamp: OOB cols never stored
  const __bf16* wRow = W + (size_t)wr * K + seg;

  auto stage = [&](int buf, int k0){
    unsigned la = (unsigned)(uintptr_t)&As[buf][ldr][seg];
    unsigned lw = (unsigned)(uintptr_t)&Ws[buf][ldr][seg];
#pragma unroll
    for (int u = 0; u < 4; u++){
      async_b128(la + 16u * u, aRow + k0 + 8 * u);
      async_b128(lw + 16u * u, wRow + k0 + 8 * u);
    }
  };

  stage(0, 0);
  wait_async0();
  __syncthreads();

  int buf = 0;
  for (int k0 = 0; k0 < K; k0 += 64){
    if (k0 + 64 < K) stage(buf ^ 1, k0 + 64);
#pragma unroll
    for (int kk = 0; kk < 2; kk++){
      bf16x16 bfr[2];
#pragma unroll
      for (int fn = 0; fn < 2; fn++)
        bfr[fn] = frag_b<72>(Ws[buf], wn + fn * 16, kk * 32, g, r);
#pragma unroll
      for (int fm = 0; fm < 4; fm++){
        bf16x16 afr = frag_a<72>(As[buf], wm + fm * 16, kk * 32, g, r);
#pragma unroll
        for (int fn = 0; fn < 2; fn++)
          acc[fm][fn] = wmma_bf16(afr, bfr[fn], acc[fm][fn]);
      }
    }
    wait_async0();
    __syncthreads();
    buf ^= 1;
  }

#pragma unroll
  for (int fm = 0; fm < 4; fm++)
#pragma unroll
    for (int fn = 0; fn < 2; fn++)
      for (int v = 0; v < 8; v++){
        int mi = m0 + wm + fm * 16 + v + 8 * g;
        int ni = n0 + wn + fn * 16 + r;
        if (ni >= N) continue;
        float c = acc[fm][fn][v];
        size_t idx = (size_t)mi * N + ni;
        if constexpr (EPI == 0)      ((float*)outp)[idx] = c;
        else if constexpr (EPI == 1) ((__bf16*)outp)[idx] = f2bf(silu_f(c));
        else if constexpr (EPI == 2) ((__bf16*)outp)[idx] = f2bf(c * bf2f(aux[idx]));
        else                         ((float*)outp)[idx] = c + res[idx];
      }
}

// ---------------- SSD intra-chunk: per (b,chunk,head) workgroup ----------------
__global__ __launch_bounds__(256)
void ssd_intra_kernel(const float* __restrict__ xbc, const float* __restrict__ cum,
                      float* __restrict__ y, float* __restrict__ Sout){
  __shared__ alignas(16) __bf16 Cs[128][72];
  __shared__ alignas(16) __bf16 Bs[128][72];
  __shared__ alignas(16) __bf16 Xt[64][136];   // X^T : [p][j]
  __shared__ alignas(16) __bf16 Ss[128][136];  // scores / BdT
  __shared__ float cumS[128];

  const int bid = blockIdx.x;
  const int h = bid & 31, cc = (bid >> 5) & 15, b = bid >> 9;
  const int tid = threadIdx.x;
  const int w = tid >> 5, lane = tid & 31, g = lane >> 4, r = lane & 15;
  const size_t rowBase = (size_t)(b * SEQ + cc * CHK) * CONVD;
  const float* xP = xbc + rowBase + h * HD;
  const float* bP = xbc + rowBase + DINNER + h * DST;
  const float* cP = xbc + rowBase + DINNER + NH * DST + h * DST;
  const float* cumP = cum + (size_t)bid * 128;

  if (tid < 128) cumS[tid] = cumP[tid];
  for (int e = tid; e < 128 * 64; e += 256){
    int j = e >> 6, n = e & 63;
    Cs[j][n] = f2bf(cP[(size_t)j * CONVD + n]);
    Bs[j][n] = f2bf(bP[(size_t)j * CONVD + n]);
    Xt[n][j] = f2bf(xP[(size_t)j * CONVD + n]);
  }
  __syncthreads();

  // Phase 1: scores(i,j) = tril(exp(cum_i - cum_j) * (C_i . B_j))  -> Ss (bf16)
  {
    const int wr2 = w >> 1, wc2 = w & 1;   // 4x2 wave grid: 32x64 tiles
    f32x8 acc[2][4];
#pragma unroll
    for (int i = 0; i < 2; i++) for (int j = 0; j < 4; j++) acc[i][j] = 0.f;
#pragma unroll
    for (int kk = 0; kk < 2; kk++){
      bf16x16 bfr[4];
#pragma unroll
      for (int fn = 0; fn < 4; fn++) bfr[fn] = frag_b<72>(Bs, wc2 * 64 + fn * 16, kk * 32, g, r);
#pragma unroll
      for (int fm = 0; fm < 2; fm++){
        bf16x16 afr = frag_a<72>(Cs, wr2 * 32 + fm * 16, kk * 32, g, r);
#pragma unroll
        for (int fn = 0; fn < 4; fn++) acc[fm][fn] = wmma_bf16(afr, bfr[fn], acc[fm][fn]);
      }
    }
#pragma unroll
    for (int fm = 0; fm < 2; fm++)
#pragma unroll
      for (int fn = 0; fn < 4; fn++)
        for (int v = 0; v < 8; v++){
          int i = wr2 * 32 + fm * 16 + v + 8 * g;
          int j = wc2 * 64 + fn * 16 + r;
          float val = (j <= i) ? acc[fm][fn][v] * __expf(cumS[i] - cumS[j]) : 0.f;
          Ss[i][j] = f2bf(val);
        }
  }
  __syncthreads();

  // Phase 2: y_intra = scores @ X  (128x64), K = 128
  {
    const int m0 = w * 16;
    f32x8 acc[4];
#pragma unroll
    for (int i = 0; i < 4; i++) acc[i] = 0.f;
#pragma unroll
    for (int kk = 0; kk < 4; kk++){
      bf16x16 afr = frag_a<136>(Ss, m0, kk * 32, g, r);
#pragma unroll
      for (int fn = 0; fn < 4; fn++){
        bf16x16 bfr = frag_b<136>(Xt, fn * 16, kk * 32, g, r);
        acc[fn] = wmma_bf16(afr, bfr, acc[fn]);
      }
    }
    float* yP = y + (size_t)(b * SEQ + cc * CHK) * DINNER + h * HD;
#pragma unroll
    for (int fn = 0; fn < 4; fn++)
      for (int v = 0; v < 8; v++){
        int i = m0 + v + 8 * g;
        int p = fn * 16 + r;
        yP[(size_t)i * DINNER + p] = acc[fn][v];
      }
  }
  __syncthreads();

  // Phase 3: S(n,p) = sum_j decay_end(j) B(j,n) X(j,p); BdT reuses Ss rows 0..63
  for (int e = tid; e < 64 * 128; e += 256){
    int n = e & 63, j = e >> 6;
    Ss[n][j] = f2bf(bf2f(Bs[j][n]) * __expf(cumS[127] - cumS[j]));
  }
  __syncthreads();
  {
    const int m0 = (w >> 1) * 16, nb = (w & 1) * 32;
    f32x8 acc[2]; acc[0] = 0.f; acc[1] = 0.f;
#pragma unroll
    for (int kk = 0; kk < 4; kk++){
      bf16x16 afr = frag_a<136>(Ss, m0, kk * 32, g, r);
#pragma unroll
      for (int fn = 0; fn < 2; fn++){
        bf16x16 bfr = frag_b<136>(Xt, nb + fn * 16, kk * 32, g, r);
        acc[fn] = wmma_bf16(afr, bfr, acc[fn]);
      }
    }
    float* sP = Sout + (size_t)bid * DST * HD;
#pragma unroll
    for (int fn = 0; fn < 2; fn++)
      for (int v = 0; v < 8; v++){
        int n = m0 + v + 8 * g;
        int p = nb + fn * 16 + r;
        sP[n * HD + p] = acc[fn][v];
      }
  }
}

// ---------------- sequential inter-chunk recurrence: per (b,h) ----------------
__global__ __launch_bounds__(256)
void chunk_rec_kernel(const float* __restrict__ cum, const float* __restrict__ S,
                      float* __restrict__ Hp){
  const int b = blockIdx.x >> 5, h = blockIdx.x & 31;
  float st[16];
#pragma unroll
  for (int u = 0; u < 16; u++) st[u] = 0.f;
  for (int cc = 0; cc < NCHK; cc++){
    size_t cbid = (size_t)(b * NCHK + cc) * NH + h;
    float A = __expf(cum[cbid * 128 + 127]);
    size_t base = cbid * DST * HD;
#pragma unroll
    for (int u = 0; u < 16; u++){
      int e = threadIdx.x + u * 256;
      Hp[base + e] = st[u];
      st[u] = A * st[u] + S[base + e];
    }
  }
}

// ---------------- y_inter + D*x + z-gating, per (b,chunk,head) ----------------
__global__ __launch_bounds__(256)
void ssd_inter_kernel(const float* __restrict__ xbc, const float* __restrict__ cum,
                      const float* __restrict__ Hp, const float* __restrict__ y,
                      const float* __restrict__ zx, const float* __restrict__ zb,
                      const float* __restrict__ Dv, __bf16* __restrict__ ybf){
  __shared__ alignas(16) __bf16 Cd[128][72];   // exp(cum_i) * C
  __shared__ alignas(16) __bf16 Ht[64][72];    // Hprev^T : [p][n]
  const int bid = blockIdx.x;
  const int h = bid & 31, cc = (bid >> 5) & 15, b = bid >> 9;
  const int tid = threadIdx.x;
  const int w = tid >> 5, lane = tid & 31, g = lane >> 4, r = lane & 15;
  const size_t rowBase = (size_t)(b * SEQ + cc * CHK) * CONVD;
  const float* cP = xbc + rowBase + DINNER + NH * DST + h * DST;
  const float* cumP = cum + (size_t)bid * 128;
  const float* hP = Hp + (size_t)bid * DST * HD;

  for (int e = tid; e < 128 * 64; e += 256){
    int i = e >> 6, n = e & 63;
    Cd[i][n] = f2bf(cP[(size_t)i * CONVD + n] * __expf(cumP[i]));
  }
  for (int e = tid; e < 64 * 64; e += 256){
    int n2 = e >> 6, p = e & 63;
    Ht[p][n2] = f2bf(hP[n2 * HD + p]);
  }
  __syncthreads();

  const int m0 = w * 16;
  f32x8 acc[4];
#pragma unroll
  for (int i = 0; i < 4; i++) acc[i] = 0.f;
#pragma unroll
  for (int kk = 0; kk < 2; kk++){
    bf16x16 afr = frag_a<72>(Cd, m0, kk * 32, g, r);
#pragma unroll
    for (int fn = 0; fn < 4; fn++){
      bf16x16 bfr = frag_b<72>(Ht, fn * 16, kk * 32, g, r);
      acc[fn] = wmma_bf16(afr, bfr, acc[fn]);
    }
  }
  const float Dh = Dv[h];
  const size_t prow = (size_t)(b * SEQ + cc * CHK);
#pragma unroll
  for (int fn = 0; fn < 4; fn++)
    for (int v = 0; v < 8; v++){
      int i = m0 + v + 8 * g;
      int p = fn * 16 + r;
      size_t pos = prow + i;
      int gcol = h * HD + p;
      float yv = y[pos * DINNER + gcol] + acc[fn][v] + Dh * xbc[pos * CONVD + gcol];
      float zv = zx[pos * DINPROJ + gcol] + zb[gcol];
      ybf[pos * DINNER + gcol] = f2bf(yv * silu_f(zv));
    }
}

// ---------------- launch ----------------
extern "C" void kernel_launch(void* const* d_in, const int* in_sizes, int n_in,
                              void* d_out, int out_size, void* d_ws, size_t ws_size,
                              hipStream_t stream){
  (void)in_sizes; (void)n_in; (void)out_size; (void)ws_size;
  const float* hs         = (const float*)d_in[0];   // [2,2048,2048]
  const float* in_proj_w  = (const float*)d_in[1];   // [8224,2048]
  const float* z_bias     = (const float*)d_in[2];   // [2048]
  const float* conv_w     = (const float*)d_in[3];   // [6144,4]
  const float* conv_b     = (const float*)d_in[4];   // [6144]
  const float* Dv         = (const float*)d_in[5];   // [32]
  const float* out_proj_w = (const float*)d_in[6];   // [2048,2048]
  const float* n1w        = (const float*)d_in[7];
  const float* n2w        = (const float*)d_in[8];
  const float* gate_w     = (const float*)d_in[9];   // [8192,2048]
  const float* up_w       = (const float*)d_in[10];  // [8192,2048]
  const float* down_w     = (const float*)d_in[11];  // [2048,8192]
  float* out = (float*)d_out;

  char* ws = (char*)d_ws;
  __bf16* xn  = (__bf16*)(ws + OFF_XN);
  __bf16* wbf = (__bf16*)(ws + OFF_WBF);
  float*  zxb = (float*) (ws + OFF_ZX);
  float*  xbc = (float*) (ws + OFF_XBC);
  float*  cum = (float*) (ws + OFF_CUM);
  float*  Sb  = (float*) (ws + OFF_S);
  float*  Hp  = (float*) (ws + OFF_HP);
  float*  yb  = (float*) (ws + OFF_Y);
  __bf16* ybf = (__bf16*)(ws + OFF_YBF);
  float*  h1  = (float*) (ws + OFF_H1);
  __bf16* h2n = (__bf16*)(ws + OFF_H2N);
  __bf16* G   = (__bf16*)(ws + OFF_G);

  // 1. rmsnorm1 -> bf16
  rmsnorm_kernel<<<MTOT, 256, 0, stream>>>(hs, n1w, xn, DMODEL);

  // 2. in_proj GEMM (fp32 out)
  {
    int n = DINPROJ * DMODEL;
    f2b_kernel<<<(n + 255) / 256, 256, 0, stream>>>(in_proj_w, wbf, n);
    gemm_bf16_kernel<0><<<dim3((DINPROJ + 127) / 128, MTOT / 128), 256, 0, stream>>>(
        xn, wbf, zxb, MTOT, DINPROJ, DMODEL, nullptr, nullptr);
  }

  // 3. causal depthwise conv
  conv_kernel<<<(BATCH * SEQ * CONVD) / 256, 256, 0, stream>>>(zxb, conv_w, conv_b, xbc);

  // 4. cumsum of -softplus(dt) per chunk/head
  cumsum_kernel<<<BATCH * NCHK * NH, 128, 0, stream>>>(zxb, cum);

  // 5. SSD intra-chunk (WMMA) -> y_intra, chunk states S
  ssd_intra_kernel<<<BATCH * NCHK * NH, 256, 0, stream>>>(xbc, cum, yb, Sb);

  // 6. sequential inter-chunk recurrence -> Hprev
  chunk_rec_kernel<<<BATCH * NH, 256, 0, stream>>>(cum, Sb, Hp);

  // 7. y_inter + D*x + silu(z)-gating -> ybf (bf16)
  ssd_inter_kernel<<<BATCH * NCHK * NH, 256, 0, stream>>>(xbc, cum, Hp, yb, zxb,
                                                          z_bias, Dv, ybf);

  // 8. out_proj GEMM + residual(hs) -> h1
  {
    int n = DMODEL * DINNER;
    f2b_kernel<<<(n + 255) / 256, 256, 0, stream>>>(out_proj_w, wbf, n);
    gemm_bf16_kernel<3><<<dim3(DMODEL / 128, MTOT / 128), 256, 0, stream>>>(
        ybf, wbf, h1, MTOT, DMODEL, DINNER, hs, nullptr);
  }

  // 9. rmsnorm2 -> bf16
  rmsnorm_kernel<<<MTOT, 256, 0, stream>>>(h1, n2w, h2n, DMODEL);

  // 10. gate GEMM + silu -> G (bf16)
  {
    int n = DFF * DMODEL;
    f2b_kernel<<<(n + 255) / 256, 256, 0, stream>>>(gate_w, wbf, n);
    gemm_bf16_kernel<1><<<dim3(DFF / 128, MTOT / 128), 256, 0, stream>>>(
        h2n, wbf, G, MTOT, DFF, DMODEL, nullptr, nullptr);
  }

  // 11. up GEMM * G -> G (bf16)
  {
    int n = DFF * DMODEL;
    f2b_kernel<<<(n + 255) / 256, 256, 0, stream>>>(up_w, wbf, n);
    gemm_bf16_kernel<2><<<dim3(DFF / 128, MTOT / 128), 256, 0, stream>>>(
        h2n, wbf, G, MTOT, DFF, DMODEL, nullptr, G);
  }

  // 12. down GEMM + residual(h1) -> out
  {
    int n = DMODEL * DFF;
    f2b_kernel<<<(n + 255) / 256, 256, 0, stream>>>(down_w, wbf, n);
    gemm_bf16_kernel<3><<<dim3(DMODEL / 128, MTOT / 128), 256, 0, stream>>>(
        G, wbf, out, MTOT, DMODEL, DFF, h1, nullptr);
  }
}
